// FNOPredictor_9088150798750
// MI455X (gfx1250) — matrix operand order repaired and checked by reading, not compiled, not run on hardware
//
#include <hip/hip_runtime.h>
#include <math.h>

// ---------------------------------------------------------------------------
// CDNA5 (gfx1250) wave32 WMMA types
// ---------------------------------------------------------------------------
typedef __bf16 v16bf __attribute__((ext_vector_type(16)));
typedef float  v8f   __attribute__((ext_vector_type(8)));

union Frag {
  uint4  q[2];
  v16bf  v;
  __bf16 h[16];
};

union Pk2 {
  __bf16   h[2];
  unsigned u;
};

__device__ __forceinline__ v8f wmma_bf16(v16bf a, v16bf b, v8f c) {
  // D(16x16 f32) = A(16x32 bf16) * B(32x16 bf16) + C
  return __builtin_amdgcn_wmma_f32_16x16x32_bf16(false, a, false, b,
                                                 (short)0, c, false, false);
}

__device__ __forceinline__ float gelu_tanh(float x) {
  float x3 = x * x * x;
  return 0.5f * x * (1.0f + tanhf(0.7978845608028654f * (x + 0.044715f * x3)));
}

// wave32 operand packings used by this code (contiguous k, same order in A and B):
//  A (16x32, 16-bit): lane l, elem j -> row = l%16, K = j + 16*(l/16)
//  B (32x16, 16-bit): lane l, elem j -> col = l%16, K = j + 16*(l/16)
//  C/D (16x16 f32):   lane l, vgpr g -> row = g + 8*(l/16), col = l%16

// ---------------------------------------------------------------------------
// Kernel 1: generate truncated-DFT bases, pre-swizzled into A-operand layout.
//   fwdR/fwdI : [64 e-steps][32 lanes][16]   A = F^T tile  (16 modes x 32 e)
//   invA      : [128 e-tiles][32 lanes][16]  A = [Cr|Ci]   (16 e-rows x 32 K)
// ---------------------------------------------------------------------------
__global__ __launch_bounds__(256)
void fno_basis_kernel(__bf16* __restrict__ fwdR, __bf16* __restrict__ fwdI,
                      __bf16* __restrict__ invA) {
  const int idx = blockIdx.x * 256 + threadIdx.x;
  const float w = 6.283185307179586f / 2048.0f;
  if (idx < 65536) {
    const int a    = idx >> 15;   // 0: real part, 1: imag part
    const int r    = idx & 32767;
    const int j    = r & 15;
    const int lane = (r >> 4) & 31;
    const int s    = r >> 9;      // e-step
    const int m    = lane & 15;   // mode (A row)
    const int k    = j + 16 * (lane >> 4);
    const int e    = s * 32 + k;
    const float th = w * (float)(m * e);
    const float val = a ? -sinf(th) : cosf(th);   // exp(-i*th)
    (a ? fwdI : fwdR)[r] = (__bf16)val;
  } else {
    const int r    = idx - 65536;
    const int j    = r & 15;
    const int lane = (r >> 4) & 31;
    const int tb   = r >> 9;      // e-tile
    const int row  = lane & 15;   // e row within tile (A row)
    const int kk   = j + 16 * (lane >> 4);
    const int e    = tb * 16 + row;
    float val;
    if (kk < 16) {                // cosine block (multiplies Yr)
      const float sc = (kk == 0 ? 1.0f : 2.0f) / 2048.0f;
      val = sc * cosf(w * (float)(kk * e));
    } else {                      // sine block (multiplies Yi)
      const int m = kk - 16;
      const float sc = (m == 0 ? 1.0f : 2.0f) / 2048.0f;
      val = -sc * sinf(w * (float)(m * e));
    }
    invA[r] = (__bf16)val;
  }
}

// ---------------------------------------------------------------------------
// Kernel 2a: pack A (MxK f32, row-major) -> bf16 tiles [mtile][ktile][r(64)][k(32)]
// ---------------------------------------------------------------------------
__global__ __launch_bounds__(256)
void pack_a_bf16(const float* __restrict__ A, __bf16* __restrict__ Ap,
                 int M, int K) {
  const int idx = blockIdx.x * 256 + threadIdx.x;
  if (idx >= M * K) return;
  const int m = idx / K;
  const int k = idx - m * K;
  const size_t dst = ((size_t)(m >> 6) * (K >> 5) + (k >> 5)) * 2048 +
                     (size_t)(m & 63) * 32 + (k & 31);
  Ap[dst] = (__bf16)A[idx];
}

// ---------------------------------------------------------------------------
// Kernel 2b: pack B (KxN f32, row-major) -> bf16 tiles [ktile][ntile][n(64)][k(32)]
// LDS transpose so both global read and write are coalesced.
// ---------------------------------------------------------------------------
__global__ __launch_bounds__(256)
void pack_b_bf16(const float* __restrict__ B, __bf16* __restrict__ Bp,
                 int K, int N) {
  __shared__ __bf16 tile[64 * 32];
  const int tid = threadIdx.x;
  const int tn = blockIdx.x;       // n-tile (64)
  const int tk = blockIdx.y;       // k-tile (32)
  for (int s = tid; s < 2048; s += 256) {
    const int k = s >> 6;
    const int n = s & 63;
    tile[n * 32 + k] = (__bf16)B[(size_t)(tk * 32 + k) * N + tn * 64 + n];
  }
  __syncthreads();
  uint4* dst = (uint4*)(Bp + ((size_t)tk * (N >> 6) + tn) * 2048);
  dst[tid] = ((const uint4*)tile)[tid];
}

// ---------------------------------------------------------------------------
// Kernel 3: WMMA GEMM on pre-packed bf16 tiles, 64x128 block tile,
// async-load-to-LDS staged, double-buffered on ASYNCcnt.
// C(MxN) f32 = Ap * Bp + bias[n].  N must be a multiple of 128.
// ---------------------------------------------------------------------------
__global__ __launch_bounds__(256)
void gemm_bias_packed(const __bf16* __restrict__ Ap, const __bf16* __restrict__ Bp,
                      const float* __restrict__ bias, float* __restrict__ C,
                      int M, int N, int K) {
  __shared__ __bf16 As[2][64 * 32];    // [row][k]
  __shared__ __bf16 Bs[2][128 * 32];   // [n][k]  (two contiguous packed n-tiles)
  const int tid  = threadIdx.x;
  const int lane = tid & 31;
  const int wid  = tid >> 5;
  const int m0 = blockIdx.y * 64;
  const int n0 = blockIdx.x * 128;
  const int rowBlk = (wid & 3) * 16;
  const int colBlk = (wid >> 2) * 64;
  const int hh  = lane >> 4;
  const int l15 = lane & 15;

  const int ktiles = K >> 5;
  const int ntiles = N >> 6;
  const __bf16* Abase = Ap + (size_t)(m0 >> 6) * ktiles * 2048;
  const __bf16* Bbase = Bp + (size_t)(n0 >> 6) * 2048;

  auto issue = [&](int tk, int buf) {
    // A: 4KB (one 16B async per lane), B: 8KB (two 16B asyncs per lane)
    unsigned la = (unsigned)(uintptr_t)&As[buf][0] + (unsigned)tid * 16u;
    unsigned lb = (unsigned)(uintptr_t)&Bs[buf][0] + (unsigned)tid * 16u;
    unsigned long long ga = (unsigned long long)(uintptr_t)(Abase + (size_t)tk * 2048)
                          + (unsigned long long)tid * 16u;
    unsigned long long gb = (unsigned long long)(uintptr_t)(Bbase + (size_t)tk * ntiles * 2048)
                          + (unsigned long long)tid * 16u;
    asm volatile("global_load_async_to_lds_b128 %0, %3, off\n\t"
                 "global_load_async_to_lds_b128 %1, %4, off\n\t"
                 "global_load_async_to_lds_b128 %2, %5, off"
                 :: "v"(la), "v"(lb), "v"((unsigned)(lb + 4096u)),
                    "v"(ga), "v"(gb), "v"((unsigned long long)(gb + 4096ull))
                 : "memory");
  };

  v8f acc[4] = {{}, {}, {}, {}};

  auto compute = [&](int buf) {
    // load ALL fragments into distinct registers first so the ten
    // ds_load_b128s pipeline, then issue the four WMMAs back-to-back
    Frag a;
    const int r = rowBlk + l15;
    a.q[0] = *(const uint4*)&As[buf][r * 32 + 16 * hh];
    a.q[1] = *(const uint4*)&As[buf][r * 32 + 16 * hh + 8];
    Frag b[4];
#pragma unroll
    for (int nt = 0; nt < 4; ++nt) {
      const int n = colBlk + nt * 16 + l15;
      b[nt].q[0] = *(const uint4*)&Bs[buf][n * 32 + 16 * hh];
      b[nt].q[1] = *(const uint4*)&Bs[buf][n * 32 + 16 * hh + 8];
    }
#pragma unroll
    for (int nt = 0; nt < 4; ++nt)
      acc[nt] = wmma_bf16(a.v, b[nt].v, acc[nt]);
  };

  issue(0, 0);
  int tk = 0;
  for (; tk + 1 < ktiles; ++tk) {
    issue(tk + 1, (tk + 1) & 1);
    asm volatile("s_wait_asynccnt 0x3" ::: "memory");   // current tile landed
    __syncthreads();
    compute(tk & 1);
    __syncthreads();
  }
  asm volatile("s_wait_asynccnt 0x0" ::: "memory");
  __syncthreads();
  compute(tk & 1);

  const int r = m0 + rowBlk + 8 * hh;
#pragma unroll
  for (int nt = 0; nt < 4; ++nt) {
    const int c = n0 + colBlk + nt * 16 + l15;
    const float bc = bias[c];
#pragma unroll
    for (int g = 0; g < 8; ++g)
      C[(size_t)(r + g) * N + c] = acc[nt][g] + bc;
  }
}

// ---------------------------------------------------------------------------
// Kernel 4: fused 4-layer FNO stack. One block (512 threads = 16 waves) per
// token t; v kept in LDS in BOTH [e][c] (pointwise A-operand) and [c][e]
// (fwd-DFT B-operand) layouts so every WMMA operand is two ds_load_b128s.
// ~273 KB LDS (320 KB WGP budget). Reads H (f32), writes Hp (packed bf16).
// ---------------------------------------------------------------------------
__global__ __launch_bounds__(512)
void fno_layers_kernel(const float* __restrict__ H, __bf16* __restrict__ Hp,
                       const __bf16* __restrict__ fwdR,
                       const __bf16* __restrict__ fwdI,
                       const __bf16* __restrict__ invA,
                       const float* __restrict__ fc0_w, const float* __restrict__ fc0_b,
                       const float* __restrict__ spec_wr, const float* __restrict__ spec_wi,
                       const float* __restrict__ pw_w, const float* __restrict__ pw_b,
                       const float* __restrict__ fc1_w, const float* __restrict__ fc1_b) {
  __shared__ __bf16 lv [2048 * 32];  // v[e][c]   128 KB
  __shared__ __bf16 lvT[32 * 2048];  // v[c][e]   128 KB (transposed copy)
  __shared__ float  lu[2048];        // token state u[e]
  __shared__ float  lXr[16 * 32];    // forward DFT Re, [m][c]
  __shared__ float  lXi[16 * 32];    // forward DFT Im
  __shared__ __bf16 lY[32 * 32];     // inverse B operand, [c][k] (k<16:Re, k>=16:Im)
  __shared__ __bf16 lpw[32 * 32];    // pointwise B operand, [cout][cin]
  __shared__ float  lpwb[32];
  __shared__ float  lw0[32], lb0[32], lw1[32];
  __shared__ float  lb1s;

  const int t    = blockIdx.x;
  const int tid  = threadIdx.x;
  const int lane = tid & 31;
  const int wid  = tid >> 5;        // 0..15
  const int hh   = lane >> 4;
  const int l15  = lane & 15;

  for (int i = tid; i < 2048; i += 512) lu[i] = H[(size_t)t * 2048 + i];

  for (int layer = 0; layer < 4; ++layer) {
    // ---- per-layer parameters into LDS ----
    if (tid < 32) {
      lw0[tid]  = fc0_w[layer * 32 + tid];
      lb0[tid]  = fc0_b[layer * 32 + tid];
      lw1[tid]  = fc1_w[layer * 32 + tid];
      lpwb[tid] = pw_b[layer * 32 + tid];
    }
    if (tid == 0) lb1s = fc1_b[layer];
    for (int i = tid; i < 1024; i += 512) {
      const int cin = i >> 5, cout = i & 31;
      lpw[cout * 32 + cin] = (__bf16)pw_w[layer * 1024 + i];
    }
    if (tid < 512) { lXr[tid] = 0.0f; lXi[tid] = 0.0f; }
    __syncthreads();

    // ---- fc0 channel lift, writing both layouts with packed b32 stores ----
    for (int p = tid; p < 1024; p += 512) {
      const int e = p << 1;
      const float u0 = lu[e], u1 = lu[e + 1];
#pragma unroll
      for (int c = 0; c < 32; c += 2) {
        const float a0 = u0 * lw0[c]     + lb0[c];
        const float a1 = u0 * lw0[c + 1] + lb0[c + 1];
        const float b0f = u1 * lw0[c]     + lb0[c];
        const float b1f = u1 * lw0[c + 1] + lb0[c + 1];
        Pk2 r0; r0.h[0] = (__bf16)a0;  r0.h[1] = (__bf16)a1;
        Pk2 r1; r1.h[0] = (__bf16)b0f; r1.h[1] = (__bf16)b1f;
        *(unsigned*)&lv[e * 32 + c]       = r0.u;
        *(unsigned*)&lv[(e + 1) * 32 + c] = r1.u;
        Pk2 t0; t0.h[0] = (__bf16)a0;  t0.h[1] = (__bf16)b0f;
        Pk2 t1; t1.h[0] = (__bf16)a1;  t1.h[1] = (__bf16)b1f;
        *(unsigned*)&lvT[c * 2048 + e]       = t0.u;
        *(unsigned*)&lvT[(c + 1) * 2048 + e] = t1.u;
      }
    }
    __syncthreads();

    // ---- forward truncated DFT: X(16x32) = F^H v, K=2048 over 16 waves ----
    {
      v8f aR0 = {}, aR1 = {}, aI0 = {}, aI1 = {};
      for (int s = wid * 4; s < wid * 4 + 4; ++s) {
        const int e0 = s * 32;
        Frag ar, ai;
        const uint4* pr = (const uint4*)fwdR + (size_t)(s * 32 + lane) * 2;
        const uint4* pi = (const uint4*)fwdI + (size_t)(s * 32 + lane) * 2;
        ar.q[0] = pr[0]; ar.q[1] = pr[1];
        ai.q[0] = pi[0]; ai.q[1] = pi[1];
        Frag b0, b1;   // B operand from transposed copy: pure b128 loads
        b0.q[0] = *(const uint4*)&lvT[l15 * 2048 + e0 + 16 * hh];
        b0.q[1] = *(const uint4*)&lvT[l15 * 2048 + e0 + 16 * hh + 8];
        b1.q[0] = *(const uint4*)&lvT[(16 + l15) * 2048 + e0 + 16 * hh];
        b1.q[1] = *(const uint4*)&lvT[(16 + l15) * 2048 + e0 + 16 * hh + 8];
        aR0 = wmma_bf16(ar.v, b0.v, aR0);
        aR1 = wmma_bf16(ar.v, b1.v, aR1);
        aI0 = wmma_bf16(ai.v, b0.v, aI0);
        aI1 = wmma_bf16(ai.v, b1.v, aI1);
      }
#pragma unroll
      for (int g = 0; g < 8; ++g) {        // cross-wave reduce via ds_add_f32
        const int m = g + 8 * hh;
        atomicAdd(&lXr[m * 32 + l15],      aR0[g]);
        atomicAdd(&lXr[m * 32 + 16 + l15], aR1[g]);
        atomicAdd(&lXi[m * 32 + l15],      aI0[g]);
        atomicAdd(&lXi[m * 32 + 16 + l15], aI1[g]);
      }
    }
    __syncthreads();

    // ---- spectral mixing: per-mode complex 32x32 (one output per thread) ----
    if (tid < 512) {
      const int m = tid >> 5, d = tid & 31;
      float yr = 0.0f, yi = 0.0f;
      const float* wr = spec_wr + (size_t)layer * 16384 + d * 16 + m;
      const float* wi = spec_wi + (size_t)layer * 16384 + d * 16 + m;
#pragma unroll 4
      for (int c = 0; c < 32; ++c) {
        const float xr = lXr[m * 32 + c];
        const float xi = lXi[m * 32 + c];
        const float wrv = wr[(size_t)c * 512];
        const float wiv = wi[(size_t)c * 512];
        yr += xr * wrv - xi * wiv;
        yi += xr * wiv + xi * wrv;
      }
      lY[d * 32 + m]      = (__bf16)yr;   // K rows 0..15 : Re
      lY[d * 32 + 16 + m] = (__bf16)yi;   // K rows 16..31: Im
    }
    __syncthreads();

    // ---- inverse DFT + pointwise (shared accumulator) + GELU + fc1 ----
    // Hoist the loop-invariant B operands / scalars out of the e-tile loop.
    Frag by0, by1, bp0, bp1;
    {
      const int na = l15, nb = 16 + l15;
      by0.q[0] = *(const uint4*)&lY[na * 32 + 16 * hh];
      by0.q[1] = *(const uint4*)&lY[na * 32 + 16 * hh + 8];
      by1.q[0] = *(const uint4*)&lY[nb * 32 + 16 * hh];
      by1.q[1] = *(const uint4*)&lY[nb * 32 + 16 * hh + 8];
      bp0.q[0] = *(const uint4*)&lpw[na * 32 + 16 * hh];
      bp0.q[1] = *(const uint4*)&lpw[na * 32 + 16 * hh + 8];
      bp1.q[0] = *(const uint4*)&lpw[nb * 32 + 16 * hh];
      bp1.q[1] = *(const uint4*)&lpw[nb * 32 + 16 * hh + 8];
    }
    const float pb0 = lpwb[l15], pb1 = lpwb[16 + l15];
    const float w1a = lw1[l15],  w1b = lw1[16 + l15];

    for (int tb = wid; tb < 128; tb += 16) {
      const int e0 = tb * 16;
      Frag ainv;
      const uint4* pv = (const uint4*)invA + (size_t)(tb * 32 + lane) * 2;
      ainv.q[0] = pv[0]; ainv.q[1] = pv[1];
      Frag av;
      {
        const int r = e0 + l15;
        av.q[0] = *(const uint4*)&lv[r * 32 + 16 * hh];
        av.q[1] = *(const uint4*)&lv[r * 32 + 16 * hh + 8];
      }
      v8f accA, accB;
#pragma unroll
      for (int g = 0; g < 8; ++g) { accA[g] = pb0; accB[g] = pb1; }
      accA = wmma_bf16(ainv.v, by0.v, accA);   // v1 (channels 0..15)
      accA = wmma_bf16(av.v,  bp0.v, accA);    // + v2
      accB = wmma_bf16(ainv.v, by1.v, accB);   // v1 (channels 16..31)
      accB = wmma_bf16(av.v,  bp1.v, accB);    // + v2
      float part[8];
#pragma unroll
      for (int g = 0; g < 8; ++g)
        part[g] = gelu_tanh(accA[g]) * w1a + gelu_tanh(accB[g]) * w1b;
#pragma unroll
      for (int g = 0; g < 8; ++g) {            // 16-lane butterfly: channel sum
        part[g] += __shfl_xor(part[g], 1, 32);
        part[g] += __shfl_xor(part[g], 2, 32);
        part[g] += __shfl_xor(part[g], 4, 32);
        part[g] += __shfl_xor(part[g], 8, 32);
      }
      if (l15 == 0) {
#pragma unroll
        for (int g = 0; g < 8; ++g)
          lu[e0 + 8 * hh + g] = part[g] + lb1s;
      }
    }
    __syncthreads();
  }

  // write result directly in the decoder GEMM's packed-A bf16 layout:
  // Hp[((tm*64 + tk)*64 + r)*32 + k%32],  tm=t/64, r=t%64, ktiles=64
  {
    const int tm = t >> 6, r = t & 63;
    for (int i = tid; i < 1024; i += 512) {
      const int k = i << 1;
      const int tk = k >> 5;
      Pk2 pk;
      pk.h[0] = (__bf16)lu[k];
      pk.h[1] = (__bf16)lu[k + 1];
      *(unsigned*)(Hp + (((size_t)tm * 64 + tk) * 64 + r) * 32 + (k & 31)) = pk.u;
    }
  }
}

// ---------------------------------------------------------------------------
// Host launcher
// ---------------------------------------------------------------------------
extern "C" void kernel_launch(void* const* d_in, const int* in_sizes, int n_in,
                              void* d_out, int out_size, void* d_ws, size_t ws_size,
                              hipStream_t stream) {
  (void)in_sizes; (void)n_in; (void)out_size; (void)ws_size;
  const float* x       = (const float*)d_in[0];
  const float* W_en    = (const float*)d_in[1];
  const float* b_en    = (const float*)d_in[2];
  const float* fc0_w   = (const float*)d_in[3];
  const float* fc0_b   = (const float*)d_in[4];
  const float* spec_wr = (const float*)d_in[5];
  const float* spec_wi = (const float*)d_in[6];
  const float* pw_w    = (const float*)d_in[7];
  const float* pw_b    = (const float*)d_in[8];
  const float* fc1_w   = (const float*)d_in[9];
  const float* fc1_b   = (const float*)d_in[10];
  const float* W_de    = (const float*)d_in[11];
  const float* b_de    = (const float*)d_in[12];
  float* out = (float*)d_out;

  char* ws = (char*)d_ws;
  float*  H    = (float*)ws;                      // 1024x2048 f32       (8 MiB)
  __bf16* fwdR = (__bf16*)(ws + 8388608);         // 64*32*16 bf16       (64 KiB)
  __bf16* fwdI = (__bf16*)(ws + 8454144);         // 64*32*16 bf16       (64 KiB)
  __bf16* invA = (__bf16*)(ws + 8519680);         // 128*32*16 bf16      (128 KiB)
  __bf16* Xp   = (__bf16*)(ws + 8650752);         // packed x   1024x1024 (2 MiB)
  __bf16* Wenp = (__bf16*)(ws + 10747904);        // packed W_en          (4 MiB)
  __bf16* Wdep = (__bf16*)(ws + 14942208);        // packed W_de          (4 MiB)
  __bf16* Hp   = (__bf16*)(ws + 19136512);        // packed FNO out       (4 MiB)

  fno_basis_kernel<<<dim3(512), dim3(256), 0, stream>>>(fwdR, fwdI, invA);
  pack_a_bf16<<<dim3(4096), dim3(256), 0, stream>>>(x, Xp, 1024, 1024);
  pack_b_bf16<<<dim3(32, 32), dim3(256), 0, stream>>>(W_en, Wenp, 1024, 2048);
  pack_b_bf16<<<dim3(16, 64), dim3(256), 0, stream>>>(W_de, Wdep, 2048, 1024);

  // encoder: H = x @ W_en + b_en   (M=1024, N=2048, K=1024)
  gemm_bias_packed<<<dim3(16, 16), dim3(256), 0, stream>>>(Xp, Wenp, b_en, H,
                                                           1024, 2048, 1024);
  // fused 4-layer FNO stack over 1024 independent tokens; emits packed Hp
  fno_layers_kernel<<<dim3(1024), dim3(512), 0, stream>>>(H, Hp, fwdR, fwdI, invA,
      fc0_w, fc0_b, spec_wr, spec_wi, pw_w, pw_b, fc1_w, fc1_b);
  // decoder: out = h @ W_de + b_de (M=1024, N=1024, K=2048)
  gemm_bias_packed<<<dim3(8, 16), dim3(256), 0, stream>>>(Hp, Wdep, b_de, out,
                                                          1024, 1024, 2048);
}
